// Self_Attention_12378095747498
// MI455X (gfx1250) — matrix operand order repaired and checked by reading, not compiled, not run on hardware
//
#include <hip/hip_runtime.h>
#include <hip/hip_bf16.h>

// ---------------------------------------------------------------------------
// CDNA5 / gfx1250 channel self-attention, bf16 WMMA pipeline with
// GLOBAL_LOAD_ASYNC_TO_LDS staging (ASYNCcnt) and double-buffered attention.
// D = A x B + C via v_wmma_f32_16x16x32_bf16 (wave32).
// Fragment layouts per CDNA5 ISA 7.12.2:
//   A (16x32 bf16): lanes 0-15 -> M=lane, K in {0..7, 16..23};
//                   lanes 16-31 -> M=lane-16, K in {8..15, 24..31}.
//   B (32x16 bf16): lanes 0-15 -> N=lane, K in {0..7, 16..23};
//                   lanes 16-31 -> N=lane-16, K in {8..15, 24..31}.
//   C/D (f32): VGPR r -> M = r + (lane<16 ? 0 : 8), N = lane%16.
// ---------------------------------------------------------------------------

typedef __attribute__((ext_vector_type(16))) __bf16 v16bf;
typedef __attribute__((ext_vector_type(8)))  float  v8f;
typedef __attribute__((ext_vector_type(4)))  unsigned int u32x4;

union BFrag { v16bf v; u32x4 u[2]; };

#define WMMA_BF16(A, B, C) \
  __builtin_amdgcn_wmma_f32_16x16x32_bf16(false, (A), false, (B), (short)0, (C), false, false)

__device__ __forceinline__ unsigned short f2bf(float f) {
  unsigned int u = __float_as_uint(f);
  unsigned int r = (u + 0x7FFFu + ((u >> 16) & 1u)) >> 16;   // round-to-nearest-even
  return (unsigned short)r;
}

// Async DMA: global -> LDS, 16B per lane.  LDS dest address comes from the low
// 32 bits of the generic pointer to the __shared__ element (AS3 offset).
__device__ __forceinline__ void async_b128(const void* g, unsigned lds_off) {
  asm volatile("global_load_async_to_lds_b128 %0, %1, off"
               :: "v"(lds_off), "v"((unsigned long long)(uintptr_t)g)
               : "memory");
}
__device__ __forceinline__ void wait_async0() {
  asm volatile("s_wait_asynccnt 0x0" ::: "memory");
}

#define BATCH 32
#define CIN   512
#define NSP   4096
#define KSTR  40   // padded LDS row stride (bf16 elems): 80B rows break b128 bank conflicts

// workspace layout (bytes)
#define WQKV_OFF 0u          // 192*512 bf16
#define WO_OFF   196608u     // 512*64 bf16
#define BIAS_OFF 262144u     // 192 f32
#define QKV_OFF  263168u     // 32*192*4096 bf16  (rows 0-63=q, 64-127=k, 128-191=v)
#define ATT_OFF  50594816u   // 32*64*64 bf16, stored TRANSPOSED: [d][c']
#define WA_OFF   50856960u   // 32*512*64 bf16   (Wa = wo @ attn)
#define ENE_OFF  52954112u   // 32*4*64*64 f32 partial energies
                             // end: 55051264 (~52.5 MB)

// ---------------------------------------------------------------------------
// K1: repack weights fp32 -> bf16, fuse qkv bias vector
// ---------------------------------------------------------------------------
__global__ __launch_bounds__(256) void prep_weights(
    const float* __restrict__ wq, const float* __restrict__ wk, const float* __restrict__ wv,
    const float* __restrict__ bq, const float* __restrict__ bk, const float* __restrict__ bv,
    const float* __restrict__ wo,
    unsigned short* __restrict__ Wqkv, unsigned short* __restrict__ Wo,
    float* __restrict__ biasqkv)
{
  int i = blockIdx.x * 256 + threadIdx.x;
  if (i < 192 * 512) {
    int r = i >> 9, c = i & 511;
    int grp = r >> 6, rr = r & 63;
    const float* w = (grp == 0) ? wq : (grp == 1) ? wk : wv;
    Wqkv[i] = f2bf(w[rr * 512 + c]);
  }
  int j = i - 192 * 512;
  if (j >= 0 && j < 512 * 64) Wo[j] = f2bf(wo[j]);
  int k = j - 512 * 64;
  if (k >= 0 && k < 192) {
    int grp = k >> 6, rr = k & 63;
    biasqkv[k] = ((grp == 0) ? bq : (grp == 1) ? bk : bv)[rr];
  }
}

// ---------------------------------------------------------------------------
// K2: QKV projection.  [192x512](bf16) @ [512x64n-tile](x->bf16) + bias
// grid (32 batches, 64 n-tiles), 256 thr = 8 waves; wave covers 48 rows x 32 cols.
// Weight tile staged by async DMA; x tile converted fp32->bf16 through VGPRs.
// ---------------------------------------------------------------------------
__global__ __launch_bounds__(256) void qkv_gemm(
    const float* __restrict__ x, const unsigned short* __restrict__ Wqkv,
    const float* __restrict__ biasqkv, unsigned short* __restrict__ QKV)
{
  __shared__ __align__(16) unsigned short sW [192 * KSTR];  // [m][k]
  __shared__ __align__(16) unsigned short sXT[ 64 * KSTR];  // [n][k] (transposed)
  const int b   = blockIdx.x;
  const int n0  = blockIdx.y * 64;
  const int tid = threadIdx.x;
  const int w = tid >> 5, lane = tid & 31;
  const int lh = lane >> 4, lm = lane & 15;
  const int m0  = (w & 3) * 48;   // 3 M-tiles per wave
  const int nn0 = (w >> 2) * 32;  // 2 N-tiles per wave
  const float* xb = x + (size_t)b * CIN * NSP;

  const v8f zero = {0.f, 0.f, 0.f, 0.f, 0.f, 0.f, 0.f, 0.f};
  v8f acc[3][2];
#pragma unroll
  for (int i = 0; i < 3; ++i)
#pragma unroll
    for (int j = 0; j < 2; ++j) acc[i][j] = zero;

  const int xk = tid >> 3;        // 0..31 (K row)
  const int xn = (tid & 7) * 8;   // 0..56 (n base)

  for (int k0 = 0; k0 < CIN; k0 += 32) {
    __syncthreads();
#pragma unroll
    for (int c = 0; c < 3; ++c) {                      // 768 16B async chunks of W tile
      int id = tid + c * 256;
      int row = id >> 2, c4 = (id & 3) * 8;
      async_b128(&Wqkv[row * 512 + k0 + c4],
                 (unsigned)(uintptr_t)&sW[row * KSTR + c4]);
    }
    { // x tile [32k x 64n] fp32 -> transposed bf16 [n][k]
      const float* src = xb + (size_t)(k0 + xk) * NSP + n0 + xn;
      float4 v0 = *(const float4*)(src);
      float4 v1 = *(const float4*)(src + 4);
      sXT[(xn + 0) * KSTR + xk] = f2bf(v0.x);
      sXT[(xn + 1) * KSTR + xk] = f2bf(v0.y);
      sXT[(xn + 2) * KSTR + xk] = f2bf(v0.z);
      sXT[(xn + 3) * KSTR + xk] = f2bf(v0.w);
      sXT[(xn + 4) * KSTR + xk] = f2bf(v1.x);
      sXT[(xn + 5) * KSTR + xk] = f2bf(v1.y);
      sXT[(xn + 6) * KSTR + xk] = f2bf(v1.z);
      sXT[(xn + 7) * KSTR + xk] = f2bf(v1.w);
    }
    wait_async0();
    __syncthreads();
    BFrag aF[3], bF[2];
#pragma unroll
    for (int i = 0; i < 3; ++i) {
      int base = (m0 + i * 16 + lm) * KSTR + lh * 8;
      aF[i].u[0] = *(const u32x4*)&sW[base];
      aF[i].u[1] = *(const u32x4*)&sW[base + 16];
    }
#pragma unroll
    for (int j = 0; j < 2; ++j) {
      int base = (nn0 + j * 16 + lm) * KSTR + lh * 8;
      bF[j].u[0] = *(const u32x4*)&sXT[base];
      bF[j].u[1] = *(const u32x4*)&sXT[base + 16];
    }
#pragma unroll
    for (int i = 0; i < 3; ++i)
#pragma unroll
      for (int j = 0; j < 2; ++j)
        acc[i][j] = WMMA_BF16(aF[i].v, bF[j].v, acc[i][j]);
  }

  unsigned short* outb = QKV + (size_t)b * 192 * NSP;
#pragma unroll
  for (int i = 0; i < 3; ++i)
#pragma unroll
    for (int j = 0; j < 2; ++j) {
      int col = n0 + nn0 + j * 16 + lm;
#pragma unroll
      for (int r = 0; r < 8; ++r) {
        int m = m0 + i * 16 + lh * 8 + r;
        outb[(size_t)m * NSP + col] = f2bf(acc[i][j][r] + biasqkv[m]);
      }
    }
}

// ---------------------------------------------------------------------------
// K3: partial energy = k @ q^T over a 1024-wide N slice.  grid (32, 4).
// Double-buffered async DMA of K/Q tiles overlaps with WMMA; one barrier/iter.
// Writes fp32 partial [64x64] per (batch, slice) -- summed by softmax kernel.
// ---------------------------------------------------------------------------
__global__ __launch_bounds__(256) void attn_kernel(
    const unsigned short* __restrict__ QKV, float* __restrict__ energyP)
{
  __shared__ __align__(16) unsigned short sK[2][64 * KSTR];
  __shared__ __align__(16) unsigned short sQ[2][64 * KSTR];
  const int b   = blockIdx.x;
  const int sl  = blockIdx.y;           // N slice
  const int n0  = sl * 1024;
  const int tid = threadIdx.x;
  const int w = tid >> 5, lane = tid & 31;
  const int lh = lane >> 4, lm = lane & 15;
  const int mt = w >> 1;                // M-tile 0..3
  const int nb = (w & 1) * 32;          // N-col base (2 tiles)
  const unsigned short* qb = QKV + (size_t)b * 192 * NSP;
  const unsigned short* kb = qb + (size_t)64 * NSP;

  const v8f zero = {0.f, 0.f, 0.f, 0.f, 0.f, 0.f, 0.f, 0.f};
  v8f acc[2]; acc[0] = zero; acc[1] = zero;

  const int lrow = tid >> 2;            // 0..63
  const int lcol = (tid & 3) * 8;       // 0,8,16,24
  const unsigned short* gk = kb + (size_t)lrow * NSP + n0 + lcol;
  const unsigned short* gq = qb + (size_t)lrow * NSP + n0 + lcol;
  const unsigned lK[2] = { (unsigned)(uintptr_t)&sK[0][lrow * KSTR + lcol],
                           (unsigned)(uintptr_t)&sK[1][lrow * KSTR + lcol] };
  const unsigned lQ[2] = { (unsigned)(uintptr_t)&sQ[0][lrow * KSTR + lcol],
                           (unsigned)(uintptr_t)&sQ[1][lrow * KSTR + lcol] };

  async_b128(gk, lK[0]);                // preload chunk 0
  async_b128(gq, lQ[0]);

  for (int it = 0; it < 32; ++it) {
    wait_async0();                      // own DMAs for current buffer done
    __syncthreads();                    // everyone's DMAs done / prior reads done
    if (it + 1 < 32) {                  // prefetch next chunk into other buffer
      int nx = (it + 1) & 1;
      async_b128(gk + (it + 1) * 32, lK[nx]);
      async_b128(gq + (it + 1) * 32, lQ[nx]);
    }
    const unsigned short* cK = &sK[it & 1][0];
    const unsigned short* cQ = &sQ[it & 1][0];
    BFrag aF;
    int abase = (mt * 16 + lm) * KSTR + lh * 8;
    aF.u[0] = *(const u32x4*)&cK[abase];
    aF.u[1] = *(const u32x4*)&cK[abase + 16];
#pragma unroll
    for (int j = 0; j < 2; ++j) {
      int bbase = (nb + j * 16 + lm) * KSTR + lh * 8;
      BFrag bF;
      bF.u[0] = *(const u32x4*)&cQ[bbase];
      bF.u[1] = *(const u32x4*)&cQ[bbase + 16];
      acc[j] = WMMA_BF16(aF.v, bF.v, acc[j]);
    }
  }

  float* ep = energyP + ((size_t)b * 4 + sl) * 4096;
#pragma unroll
  for (int j = 0; j < 2; ++j) {
    int col = nb + j * 16 + lm;
#pragma unroll
    for (int r = 0; r < 8; ++r)
      ep[(mt * 16 + lh * 8 + r) * 64 + col] = acc[j][r];
  }
}

// ---------------------------------------------------------------------------
// K3b: sum 4 partial energies, row softmax, emit attn^T (bf16).  grid 32 x 64thr.
// ---------------------------------------------------------------------------
__global__ __launch_bounds__(64) void softmax_kernel(
    const float* __restrict__ energyP, unsigned short* __restrict__ attnT)
{
  const int b = blockIdx.x;
  const int c = threadIdx.x;                       // row 0..63
  const float* e = energyP + (size_t)b * 4 * 4096 + c * 64;
  float mx = -1e30f;
  for (int d = 0; d < 64; ++d) {
    float s = e[d] + e[4096 + d] + e[8192 + d] + e[12288 + d];
    mx = fmaxf(mx, s);
  }
  float tot = 0.f;
  for (int d = 0; d < 64; ++d) {
    float s = e[d] + e[4096 + d] + e[8192 + d] + e[12288 + d];
    tot += __expf(s - mx);
  }
  float inv = 1.f / tot;
  unsigned short* at = attnT + b * 4096;
  for (int d = 0; d < 64; ++d) {
    float s = e[d] + e[4096 + d] + e[8192 + d] + e[12288 + d];
    at[d * 64 + c] = f2bf(__expf(s - mx) * inv);   // transposed store
  }
}

// ---------------------------------------------------------------------------
// K4a: Wa = wo @ attn   ([512x64] @ [64x64], bf16 out).  grid (32, 4).
// ---------------------------------------------------------------------------
__global__ __launch_bounds__(256) void wo_attn_kernel(
    const unsigned short* __restrict__ Wo, const unsigned short* __restrict__ attnT,
    unsigned short* __restrict__ Wa)
{
  __shared__ __align__(16) unsigned short sA[128 * KSTR];
  __shared__ __align__(16) unsigned short sB[ 64 * KSTR];
  const int b  = blockIdx.x;
  const int m0 = blockIdx.y * 128;
  const int tid = threadIdx.x;
  const int w = tid >> 5, lane = tid & 31;
  const int lh = lane >> 4, lm = lane & 15;
  const unsigned short* at = attnT + b * 4096;

  const v8f zero = {0.f, 0.f, 0.f, 0.f, 0.f, 0.f, 0.f, 0.f};
  v8f acc[4];
#pragma unroll
  for (int j = 0; j < 4; ++j) acc[j] = zero;

  for (int k0 = 0; k0 < 64; k0 += 32) {
    __syncthreads();
#pragma unroll
    for (int c = 0; c < 2; ++c) {
      int id = tid + c * 256;
      int row = id >> 2, c4 = (id & 3) * 8;
      async_b128(&Wo[(m0 + row) * 64 + k0 + c4],
                 (unsigned)(uintptr_t)&sA[row * KSTR + c4]);
    }
    {
      int row = tid >> 2, c4 = (tid & 3) * 8;
      async_b128(&at[row * 64 + k0 + c4],
                 (unsigned)(uintptr_t)&sB[row * KSTR + c4]);
    }
    wait_async0();
    __syncthreads();
    BFrag aF;
    int abase = (w * 16 + lm) * KSTR + lh * 8;
    aF.u[0] = *(const u32x4*)&sA[abase];
    aF.u[1] = *(const u32x4*)&sA[abase + 16];
#pragma unroll
    for (int j = 0; j < 4; ++j) {
      int bbase = (j * 16 + lm) * KSTR + lh * 8;
      BFrag bF;
      bF.u[0] = *(const u32x4*)&sB[bbase];
      bF.u[1] = *(const u32x4*)&sB[bbase + 16];
      acc[j] = WMMA_BF16(aF.v, bF.v, acc[j]);
    }
  }
  unsigned short* wab = Wa + ((size_t)b * 512 + m0) * 64;
#pragma unroll
  for (int j = 0; j < 4; ++j) {
    int col = j * 16 + lm;
#pragma unroll
    for (int r = 0; r < 8; ++r)
      wab[(w * 16 + lh * 8 + r) * 64 + col] = f2bf(acc[j][r]);
  }
}

// ---------------------------------------------------------------------------
// K4b: out = gamma * (Wa @ v + bo) + x.   grid (32, 2 m-blocks, 64 n-tiles).
// ---------------------------------------------------------------------------
__global__ __launch_bounds__(256) void out_gemm(
    const unsigned short* __restrict__ Wa, const unsigned short* __restrict__ QKV,
    const float* __restrict__ x, const float* __restrict__ bo,
    const float* __restrict__ gamma, float* __restrict__ out)
{
  __shared__ __align__(16) unsigned short sWa[256 * KSTR];
  __shared__ __align__(16) unsigned short sVT[ 64 * KSTR];   // v transposed [n][d]
  const int b  = blockIdx.x;
  const int m0 = blockIdx.y * 256;
  const int n0 = blockIdx.z * 64;
  const int tid = threadIdx.x;
  const int w = tid >> 5, lane = tid & 31;
  const int lh = lane >> 4, lm = lane & 15;
  const unsigned short* wab = Wa + (size_t)b * 512 * 64;
  const unsigned short* vb  = QKV + ((size_t)b * 192 + 128) * NSP;

  const v8f zero = {0.f, 0.f, 0.f, 0.f, 0.f, 0.f, 0.f, 0.f};
  v8f acc[2][4];
#pragma unroll
  for (int i = 0; i < 2; ++i)
#pragma unroll
    for (int j = 0; j < 4; ++j) acc[i][j] = zero;

  const int vd = tid >> 3;        // 0..31 (d row within chunk)
  const int vn = (tid & 7) * 8;   // 0..56

  for (int k0 = 0; k0 < 64; k0 += 32) {
    __syncthreads();
#pragma unroll
    for (int c = 0; c < 4; ++c) {
      int id = tid + c * 256;
      int row = id >> 2, c4 = (id & 3) * 8;
      async_b128(&wab[(m0 + row) * 64 + k0 + c4],
                 (unsigned)(uintptr_t)&sWa[row * KSTR + c4]);
    }
    { // v tile transpose through VGPRs (layout change, can't DMA)
      u32x4 raw = *(const u32x4*)&vb[(size_t)(k0 + vd) * NSP + n0 + vn];
      const unsigned short* rs = (const unsigned short*)&raw;
#pragma unroll
      for (int j = 0; j < 8; ++j) sVT[(vn + j) * KSTR + vd] = rs[j];
    }
    wait_async0();
    __syncthreads();
    BFrag aF[2];
#pragma unroll
    for (int i = 0; i < 2; ++i) {
      int abase = ((w * 2 + i) * 16 + lm) * KSTR + lh * 8;
      aF[i].u[0] = *(const u32x4*)&sWa[abase];
      aF[i].u[1] = *(const u32x4*)&sWa[abase + 16];
    }
#pragma unroll
    for (int j = 0; j < 4; ++j) {
      int bbase = (j * 16 + lm) * KSTR + lh * 8;
      BFrag bF;
      bF.u[0] = *(const u32x4*)&sVT[bbase];
      bF.u[1] = *(const u32x4*)&sVT[bbase + 16];
#pragma unroll
      for (int i = 0; i < 2; ++i)
        acc[i][j] = WMMA_BF16(aF[i].v, bF.v, acc[i][j]);
    }
  }
  const float g = gamma[0];
  const float* xb = x + (size_t)b * CIN * NSP;
  float* ob = out + (size_t)b * CIN * NSP;
#pragma unroll
  for (int i = 0; i < 2; ++i)
#pragma unroll
    for (int j = 0; j < 4; ++j) {
      int col = n0 + j * 16 + lm;
#pragma unroll
      for (int r = 0; r < 8; ++r) {
        int m = m0 + w * 32 + i * 16 + lh * 8 + r;
        size_t idx = (size_t)m * NSP + col;
        ob[idx] = g * (acc[i][j][r] + bo[m]) + xb[idx];
      }
    }
}

// ---------------------------------------------------------------------------
extern "C" void kernel_launch(void* const* d_in, const int* in_sizes, int n_in,
                              void* d_out, int out_size, void* d_ws, size_t ws_size,
                              hipStream_t stream)
{
  const float* x     = (const float*)d_in[0];
  const float* wq    = (const float*)d_in[1];
  const float* bq    = (const float*)d_in[2];
  const float* wk    = (const float*)d_in[3];
  const float* bk    = (const float*)d_in[4];
  const float* wv    = (const float*)d_in[5];
  const float* bv    = (const float*)d_in[6];
  const float* wo    = (const float*)d_in[7];
  const float* bo    = (const float*)d_in[8];
  const float* gamma = (const float*)d_in[9];

  char* ws = (char*)d_ws;
  unsigned short* Wqkv  = (unsigned short*)(ws + WQKV_OFF);
  unsigned short* Wo    = (unsigned short*)(ws + WO_OFF);
  float*          biasq = (float*)(ws + BIAS_OFF);
  unsigned short* QKV   = (unsigned short*)(ws + QKV_OFF);
  unsigned short* attnT = (unsigned short*)(ws + ATT_OFF);
  unsigned short* Wa    = (unsigned short*)(ws + WA_OFF);
  float*          eneP  = (float*)(ws + ENE_OFF);
  float* out = (float*)d_out;

  prep_weights<<<513, 256, 0, stream>>>(wq, wk, wv, bq, bk, bv, wo, Wqkv, Wo, biasq);
  qkv_gemm<<<dim3(32, 64), 256, 0, stream>>>(x, Wqkv, biasq, QKV);
  attn_kernel<<<dim3(32, 4), 256, 0, stream>>>(QKV, eneP);
  softmax_kernel<<<32, 64, 0, stream>>>(eneP, attnT);
  wo_attn_kernel<<<dim3(32, 4), 256, 0, stream>>>(Wo, attnT, Wa);
  out_gemm<<<dim3(32, 2, 64), 256, 0, stream>>>(Wa, QKV, x, bo, gamma, out);
}
